// MoERouter_17368847745255
// MI455X (gfx1250) — matrix operand order repaired and checked by reading, not compiled, not run on hardware
//
#include <hip/hip_runtime.h>
#include <hip/hip_bf16.h>

typedef __attribute__((ext_vector_type(16))) __bf16 v16bf;
typedef __attribute__((ext_vector_type(8)))  float  v8f;
typedef __attribute__((ext_vector_type(4)))  float  f4;

constexpr int NUM_EXPERTS = 64;
constexpr int TOP_K       = 2;
constexpr int HIDDEN      = 4096;
constexpr int NTOK        = 4 * 4096;        // 16384 tokens
constexpr int NFLAT       = NTOK * TOP_K;    // 32768
constexpr int CHUNK       = 256;
constexpr int NCHUNK      = NFLAT / CHUNK;   // 128
constexpr int KCHUNK      = 128;             // K-slab staged in LDS
constexpr int NSLAB       = HIDDEN / KCHUNK; // 32
constexpr int SLAB_BYTES  = NUM_EXPERTS * KCHUNK * 2;  // 16 KB

// ---------------------------------------------------------------------------
// K0: one-shot gate_w fp32 -> bf16 (row-major [E][H]).
// ---------------------------------------------------------------------------
__global__ __launch_bounds__(256) void k_cvt_gate(const float* __restrict__ gw,
                                                  __bf16* __restrict__ gwb, int n) {
    int i = blockIdx.x * 256 + threadIdx.x;
    if (i < n) gwb[i] = (__bf16)gw[i];
}

// ---------------------------------------------------------------------------
// K1: fused router GEMM (bf16 WMMA). Gate slabs are double-buffered in LDS
// via GLOBAL_LOAD_ASYNC_TO_LDS_B128 (ASYNCcnt), overlapping the next slab's
// fetch with the current slab's WMMAs. 8 waves/block, 16 tokens/wave.
// ---------------------------------------------------------------------------
__global__ __launch_bounds__(256) void k_router(const float*  __restrict__ hidden,
                                                const __bf16* __restrict__ gwb,
                                                float* __restrict__ out_rw,
                                                int*   __restrict__ out_ti,
                                                int*   __restrict__ ws_flat) {
    // 32 KB: two 16 KB B slabs during the K loop; reused as [8][16][64] f32
    // logit staging afterwards (barrier-separated).
    __shared__ __align__(128) char smem[2 * SLAB_BYTES];
    __bf16* bbuf = (__bf16*)smem;

    const int tid  = threadIdx.x;
    const int wave = tid >> 5;
    const int lane = tid & 31;
    const int lo   = lane & 15;        // row (A) / column (B) within tile
    const int hi   = lane >> 4;        // K-half selector
    const int tokenBase = blockIdx.x * 128 + wave * 16;

    const float* __restrict__ arow = hidden + (size_t)(tokenBase + lo) * HIDDEN;

    // Async B staging: thread t owns 64 B of the slab: expert row t/4,
    // column block (t%4)*32 halfs, copied as 4x async b128 (IOFFSET applies
    // to both the LDS and global addresses).
    const int srow = tid >> 2;              // 0..63 (expert row)
    const int scol = (tid & 3) * 32;        // 0,32,64,96 (halfs)
    const unsigned lds_my =
        (unsigned)(size_t)(bbuf + srow * KCHUNK + scol);         // buf0 LDS addr
    const unsigned long long g_my =
        (unsigned long long)(gwb + (size_t)srow * HIDDEN + scol);

    // Prologue: slab 0 -> buffer 0.
    asm volatile(
        "global_load_async_to_lds_b128 %0, %1, off\n\t"
        "global_load_async_to_lds_b128 %0, %1, off offset:16\n\t"
        "global_load_async_to_lds_b128 %0, %1, off offset:32\n\t"
        "global_load_async_to_lds_b128 %0, %1, off offset:48"
        :: "v"(lds_my), "v"(g_my) : "memory");

    v8f acc0 = {}, acc1 = {}, acc2 = {}, acc3 = {};

    for (int i = 0; i < NSLAB; ++i) {
        const int cur = i & 1;
        if (i + 1 < NSLAB) {
            // Prefetch slab i+1 into the other buffer, then wait for slab i
            // (async loads complete in order: <=4 outstanding => slab i done).
            const unsigned l = lds_my + (unsigned)((cur ^ 1) * SLAB_BYTES);
            const unsigned long long g =
                g_my + (unsigned long long)(i + 1) * (KCHUNK * 2);
            asm volatile(
                "global_load_async_to_lds_b128 %0, %1, off\n\t"
                "global_load_async_to_lds_b128 %0, %1, off offset:16\n\t"
                "global_load_async_to_lds_b128 %0, %1, off offset:32\n\t"
                "global_load_async_to_lds_b128 %0, %1, off offset:48"
                :: "v"(l), "v"(g) : "memory");
            asm volatile("s_wait_asynccnt 0x4" ::: "memory");
        } else {
            asm volatile("s_wait_asynccnt 0x0" ::: "memory");
        }
        __syncthreads();                    // all waves' slab-i portions visible

        const __bf16* slab = bbuf + cur * (NUM_EXPERTS * KCHUNK);
        const int kc = i * KCHUNK;

#pragma unroll
        for (int ks = 0; ks < KCHUNK; ks += 32) {
            const int kk = kc + ks;
            // A fragment: contiguous K order — lanes<16 hold K {0..15},
            // lanes>=16 hold K {16..31}; elements ascend K within lane.
            f4 a0 = *(const f4*)(arow + kk + 16 * hi);
            f4 a1 = *(const f4*)(arow + kk + 16 * hi + 4);
            f4 a2 = *(const f4*)(arow + kk + 16 * hi + 8);
            f4 a3 = *(const f4*)(arow + kk + 16 * hi + 12);
            v16bf a;
#pragma unroll
            for (int j = 0; j < 4; ++j) {
                a[j]      = (__bf16)a0[j];
                a[4 + j]  = (__bf16)a1[j];
                a[8 + j]  = (__bf16)a2[j];
                a[12 + j] = (__bf16)a3[j];
            }
            // B fragments from LDS: lane = column N (lo), K-half by hi;
            // 32 contiguous bytes per lane per expert tile.
            const __bf16* bp = slab + lo * KCHUNK + ks + 16 * hi;
            v16bf b0 = *(const v16bf*)(bp);
            v16bf b1 = *(const v16bf*)(bp + 16 * KCHUNK);
            v16bf b2 = *(const v16bf*)(bp + 32 * KCHUNK);
            v16bf b3 = *(const v16bf*)(bp + 48 * KCHUNK);

            acc0 = __builtin_amdgcn_wmma_f32_16x16x32_bf16(false, a, false, b0, (short)0, acc0, false, false);
            acc1 = __builtin_amdgcn_wmma_f32_16x16x32_bf16(false, a, false, b1, (short)0, acc1, false, false);
            acc2 = __builtin_amdgcn_wmma_f32_16x16x32_bf16(false, a, false, b2, (short)0, acc2, false, false);
            acc3 = __builtin_amdgcn_wmma_f32_16x16x32_bf16(false, a, false, b3, (short)0, acc3, false, false);
        }
        __syncthreads();                    // slab fully consumed before reuse
    }

    // Reuse smem as logit staging: [8 waves][16 tokens][64 experts] f32 (32 KB).
    // C/D layout: VGPR j holds M = j + 8*hi, N = lo.
    float* lgw = (float*)smem + wave * 16 * NUM_EXPERTS;
#pragma unroll
    for (int j = 0; j < 8; ++j) {
        const int m = j + 8 * hi;
        lgw[m * NUM_EXPERTS + lo]      = acc0[j];
        lgw[m * NUM_EXPERTS + 16 + lo] = acc1[j];
        lgw[m * NUM_EXPERTS + 32 + lo] = acc2[j];
        lgw[m * NUM_EXPERTS + 48 + lo] = acc3[j];
    }
    __syncthreads();

    if (hi == 0) {              // lanes 0..15: one token each
        const int token = tokenBase + lo;
        const float* row = lgw + lo * NUM_EXPERTS;

        float mx = -1e30f;
#pragma unroll 8
        for (int e = 0; e < NUM_EXPERTS; ++e) mx = fmaxf(mx, row[e]);
        float z = 0.0f;
#pragma unroll 8
        for (int e = 0; e < NUM_EXPERTS; ++e) z += __expf(row[e] - mx);

        // top-1 then top-2 (ties -> lowest index, matching lax.top_k)
        int i1 = 0; float v1 = row[0];
        for (int e = 1; e < NUM_EXPERTS; ++e)
            if (row[e] > v1) { v1 = row[e]; i1 = e; }
        int i2 = -1; float v2 = -1e30f;
        for (int e = 0; e < NUM_EXPERTS; ++e)
            if (e != i1 && row[e] > v2) { v2 = row[e]; i2 = e; }

        const float p1 = __expf(v1 - mx) / z;
        const float p2 = __expf(v2 - mx) / z;
        const float s  = p1 + p2 + 1e-8f;

        out_rw[2 * token]     = p1 / s;
        out_rw[2 * token + 1] = p2 / s;
        out_ti[2 * token]     = i1;
        out_ti[2 * token + 1] = i2;
        ws_flat[2 * token]     = i1;
        ws_flat[2 * token + 1] = i2;
    }
}

// ---------------------------------------------------------------------------
// K2: per-chunk 64-bin histogram of expert assignments.
// ---------------------------------------------------------------------------
__global__ __launch_bounds__(CHUNK) void k_hist(const int* __restrict__ flat,
                                                int* __restrict__ hist) {
    __shared__ int h[NUM_EXPERTS];
    const int t = threadIdx.x, c = blockIdx.x;
    if (t < NUM_EXPERTS) h[t] = 0;
    __syncthreads();
    atomicAdd(&h[flat[c * CHUNK + t]], 1);
    __syncthreads();
    if (t < NUM_EXPERTS) hist[c * NUM_EXPERTS + t] = h[t];
}

// ---------------------------------------------------------------------------
// K3: expert totals -> inclusive cumsum (expert_offsets) + per-(chunk,expert)
// scatter bases. Also writes the zero losses.
// ---------------------------------------------------------------------------
__global__ __launch_bounds__(NUM_EXPERTS) void k_scan(const int* __restrict__ hist,
                                                      int* __restrict__ chunk_base,
                                                      int* __restrict__ out_eo,
                                                      float* __restrict__ out_aux,
                                                      float* __restrict__ out_z) {
    __shared__ int cnt[NUM_EXPERTS], excl[NUM_EXPERTS];
    const int e = threadIdx.x;
    int tot = 0;
    for (int c = 0; c < NCHUNK; ++c) tot += hist[c * NUM_EXPERTS + e];
    cnt[e] = tot;
    __syncthreads();
    if (e == 0) {
        int run = 0;
        for (int i = 0; i < NUM_EXPERTS; ++i) {
            excl[i] = run;
            run += cnt[i];
            out_eo[i] = run;            // inclusive cumsum, as in reference
        }
        out_aux[0] = 0.0f;
        out_z[0]   = 0.0f;
    }
    __syncthreads();
    int run = excl[e];
    for (int c = 0; c < NCHUNK; ++c) {
        chunk_base[c * NUM_EXPERTS + e] = run;
        run += hist[c * NUM_EXPERTS + e];
    }
}

// ---------------------------------------------------------------------------
// K4: stable scatter — gather_indices[sorted_pos] = original_index.
// ---------------------------------------------------------------------------
__global__ __launch_bounds__(CHUNK) void k_scatter(const int* __restrict__ flat,
                                                   const int* __restrict__ chunk_base,
                                                   int* __restrict__ out_gi) {
    __shared__ int ex[CHUNK];
    const int t = threadIdx.x, c = blockIdx.x;
    const int e = flat[c * CHUNK + t];
    ex[t] = e;
    __syncthreads();
    int r = 0;
    for (int j = 0; j < t; ++j) r += (ex[j] == e);
    out_gi[chunk_base[c * NUM_EXPERTS + e] + r] = c * CHUNK + t;
}

// ---------------------------------------------------------------------------
extern "C" void kernel_launch(void* const* d_in, const int* in_sizes, int n_in,
                              void* d_out, int out_size, void* d_ws, size_t ws_size,
                              hipStream_t stream) {
    const float* hidden = (const float*)d_in[0];   // [4,4096,4096] f32
    const float* gate_w = (const float*)d_in[1];   // [64,4096]     f32

    // Output layout (flat, return order):
    float* out       = (float*)d_out;
    float* out_rw    = out;                          // 32768 f32
    int*   out_ti    = (int*)(out + NFLAT);          // 32768 i32
    int*   out_eo    = (int*)(out + 2 * NFLAT);      // 64    i32
    int*   out_gi    = (int*)(out + 2 * NFLAT + NUM_EXPERTS);  // 32768 i32
    float* out_aux   = out + 2 * NFLAT + NUM_EXPERTS + NFLAT;  // 1
    float* out_z     = out_aux + 1;                            // 1

    // Workspace layout
    char*   ws    = (char*)d_ws;
    __bf16* gwb   = (__bf16*)ws;                               // 512 KB
    int*    flat  = (int*)(ws + 512 * 1024);                   // 128 KB
    int*    hist  = (int*)(ws + 512 * 1024 + 128 * 1024);      //  32 KB
    int*    cbase = (int*)(ws + 512 * 1024 + 160 * 1024);      //  32 KB

    const int ngw = NUM_EXPERTS * HIDDEN;
    k_cvt_gate<<<(ngw + 255) / 256, 256, 0, stream>>>(gate_w, gwb, ngw);
    k_router<<<NTOK / 128, 256, 0, stream>>>(hidden, gwb, out_rw, out_ti, flat);
    k_hist<<<NCHUNK, CHUNK, 0, stream>>>(flat, hist);
    k_scan<<<1, NUM_EXPERTS, 0, stream>>>(hist, cbase, out_eo, out_aux, out_z);
    k_scatter<<<NCHUNK, CHUNK, 0, stream>>>(flat, cbase, out_gi);
}
